// FFFLayer_16673063043521
// MI455X (gfx1250) — matrix operand order repaired and checked by reading, not compiled, not run on hardware
//
#include <hip/hip_runtime.h>
#include <cmath>

// FFF tree layer: x[8192,2048] f32, in_weight[2*2047,2048], in_bias[2*2047],
// out_weight[2*2047,2048] -> out[8192,2048] f32.
// 16 tokens per block, 128 threads (4 wave32s). Gating dots = diagonal of a
// 16x16 WMMA tile (A = x tile f16 in LDS, B = gathered weight rows).

#define WIDTHW     2048
#define DEPTH      10
#define NPAR       2
#define N_NODES    2047
#define TPB        128        // 4 waves
#define TOKS       16         // tokens per block
#define XSTRIDE    2056       // halves per x row in LDS (pad 8 -> no bank alias)
#define ACCSTRIDE  2052       // floats per acc row in LDS (pad 4)

typedef __attribute__((ext_vector_type(16))) _Float16 v16h;
typedef __attribute__((ext_vector_type(8)))  float    v8f;

union HalfPack {
    v16h     v;
    uint4    u[2];
    _Float16 h[16];
};

__global__ __launch_bounds__(TPB) void fff_tree_kernel(
    const float* __restrict__ x,
    const float* __restrict__ in_w,
    const float* __restrict__ in_b,
    const float* __restrict__ out_w,
    float* __restrict__ out)
{
    extern __shared__ char smem[];
    float*    accS  = (float*)smem;                                        // 16*ACCSTRIDE f32
    _Float16* xS    = (_Float16*)(smem + (size_t)TOKS * ACCSTRIDE * 4);    // 16*XSTRIDE f16
    float*    partS = (float*)((char*)xS + (size_t)TOKS * XSTRIDE * 2);    // 4 waves * 16
    float*    gS    = partS + 64;                                          // 16
    int*      rowS  = (int*)(gS + 16);                                     // 16
    int*      nodeS = rowS + 16;                                           // 16

    const int tid  = threadIdx.x;
    const int lane = tid & 31;
    const int wv   = tid >> 5;          // wave id 0..3, owns K slice [wv*512, wv*512+512)
    const int tok0 = blockIdx.x * TOKS;

    // ---- Stage x tile (f32 -> f16) into LDS, zero the f32 accumulator tile.
    for (int n = 0; n < TOKS; ++n) {
        const float4* xr = (const float4*)(x + (size_t)(tok0 + n) * WIDTHW + tid * 16);
        _Float16* xd = xS + n * XSTRIDE + tid * 16;
        float4* ar = (float4*)(accS + n * ACCSTRIDE + tid * 16);
#pragma unroll
        for (int q = 0; q < 4; ++q) {
            float4 v = xr[q];
            xd[q * 4 + 0] = (_Float16)v.x;
            xd[q * 4 + 1] = (_Float16)v.y;
            xd[q * 4 + 2] = (_Float16)v.z;
            xd[q * 4 + 3] = (_Float16)v.w;
            ar[q] = make_float4(0.f, 0.f, 0.f, 0.f);
        }
    }

    const int half = (lane >> 4) & 1;   // lane 0-15 vs 16-31 (A/B layout halves)
    const int myn  = lane & 15;         // token / matrix column owned by this lane

    for (int p = 0; p < NPAR; ++p) {
        if (tid < TOKS) nodeS[tid] = 0;
        __syncthreads();

        for (int lvl = 0; lvl <= DEPTH; ++lvl) {
            // ---- resolve gathered rows for this level, prefetch out_weight rows
            if (tid < TOKS) {
                int r = p * N_NODES + nodeS[tid];
                rowS[tid] = r;
                __builtin_prefetch(out_w + (size_t)r * WIDTHW, 0, 1);
            }
            __syncthreads();

            // ---- Phase 1: gating dots via WMMA (diagonal of X * Wg^T).
            // A 16x32 f16 layout: lanes 0-15 hold K runs [k0,k0+8) and [k0+16,k0+24)
            // of row M=lane; lanes 16-31 hold [k0+8,..) and [k0+24,..).
            // B 32x16 f16 layout: same K runs as A, for column N=lane%16.
            const int myrow = rowS[myn];
            const _Float16* xb = xS + myn * XSTRIDE + half * 8;
            const float*    wb = in_w + (size_t)myrow * WIDTHW + half * 8;

            v8f c = {};
            for (int chunk = 0; chunk < 16; ++chunk) {
                const int k0 = wv * 512 + chunk * 32;
                HalfPack A, B;
                A.u[0] = *(const uint4*)(xb + k0);
                A.u[1] = *(const uint4*)(xb + k0 + 16);
                float t[16];
                const float4* wp = (const float4*)(wb + k0);
                *(float4*)&t[0]  = wp[0];
                *(float4*)&t[4]  = wp[1];
                *(float4*)&t[8]  = wp[4];
                *(float4*)&t[12] = wp[5];
#pragma unroll
                for (int j = 0; j < 16; ++j) B.h[j] = (_Float16)t[j];
                c = __builtin_amdgcn_wmma_f32_16x16x32_f16(
                        /*neg_a=*/false, A.v, /*neg_b=*/false, B.v,
                        /*c_mod=*/(short)0, c, /*reuse_a=*/false, /*reuse_b=*/false);
            }

            // Diagonal of C: m<8 -> VGPR m, lane m ; m>=8 -> VGPR m-8, lane m+16.
            float d[16];
#pragma unroll
            for (int i = 0; i < 8; ++i) {
                d[i]     = __int_as_float(__builtin_amdgcn_readlane(__float_as_int(c[i]), i));
                d[i + 8] = __int_as_float(__builtin_amdgcn_readlane(__float_as_int(c[i]), i + 24));
            }
            if (lane == 0) {
#pragma unroll
                for (int m = 0; m < 16; ++m) partS[wv * 16 + m] = d[m];
            }
            __syncthreads();

            // ---- Phase 2: reduce K slices, bias, exact gelu, branch.
            if (tid < TOKS) {
                float val = partS[tid] + partS[16 + tid] + partS[32 + tid] + partS[48 + tid]
                          + in_b[rowS[tid]];
                gS[tid] = 0.5f * val * (1.0f + erff(val * 0.70710678118654752f));
                nodeS[tid] = 2 * nodeS[tid] + 1 + (val > 0.0f ? 1 : 0);
            }
            __syncthreads();

            // ---- Phase 3: acc[n,:] += g[n] * out_w[row[n],:]  (coalesced, float4)
            for (int m = 0; m < TOKS; ++m) {
                const float gm = gS[m];
                const float4* wr = (const float4*)(out_w + (size_t)rowS[m] * WIDTHW + tid * 16);
                float4* ar = (float4*)(accS + m * ACCSTRIDE + tid * 16);
#pragma unroll
                for (int q = 0; q < 4; ++q) {
                    float4 w4 = wr[q];
                    float4 a4 = ar[q];
                    a4.x += gm * w4.x; a4.y += gm * w4.y;
                    a4.z += gm * w4.z; a4.w += gm * w4.w;
                    ar[q] = a4;
                }
            }
            __syncthreads();
        }
    }

    // ---- Write accumulator tile to global output (coalesced).
    for (int m = 0; m < TOKS; ++m) {
        float4* orow = (float4*)(out + (size_t)(tok0 + m) * WIDTHW + tid * 16);
        const float4* ar = (const float4*)(accS + m * ACCSTRIDE + tid * 16);
#pragma unroll
        for (int q = 0; q < 4; ++q) orow[q] = ar[q];
    }
}

extern "C" void kernel_launch(void* const* d_in, const int* in_sizes, int n_in,
                              void* d_out, int out_size, void* d_ws, size_t ws_size,
                              hipStream_t stream) {
    (void)n_in; (void)d_ws; (void)ws_size; (void)out_size;
    const float* x    = (const float*)d_in[0];
    const float* in_w = (const float*)d_in[1];
    const float* in_b = (const float*)d_in[2];
    const float* outw = (const float*)d_in[3];
    float* out = (float*)d_out;

    const int ntok   = in_sizes[0] / WIDTHW;   // 8192
    const int blocks = ntok / TOKS;            // 512

    const size_t shmem = (size_t)TOKS * ACCSTRIDE * sizeof(float)     // acc tile
                       + (size_t)TOKS * XSTRIDE * sizeof(_Float16)    // x tile (f16)
                       + 64 * sizeof(float)                           // wave partials
                       + 16 * sizeof(float)                           // gelu values
                       + 16 * sizeof(int)                             // row indices
                       + 16 * sizeof(int);                            // node state

    // CDNA5 WGP has 320KB LDS; raise the per-kernel dynamic-LDS cap (~198KB used).
    hipFuncSetAttribute((const void*)fff_tree_kernel,
                        hipFuncAttributeMaxDynamicSharedMemorySize, (int)shmem);

    fff_tree_kernel<<<blocks, TPB, shmem, stream>>>(x, in_w, in_b, outw, out);
}
